// AVWDCRNN_31602369364703
// MI455X (gfx1250) — matrix operand order repaired and checked by reading, not compiled, not run on hardware
//
#include <hip/hip_runtime.h>
#include <math.h>

#define NODE 2048
#define EDIM 16
#define HID  64
#define BB   16
#define TT   24

typedef __attribute__((ext_vector_type(16))) __bf16 v16bf;
typedef __attribute__((ext_vector_type(8)))  float  v8f;
typedef __attribute__((ext_vector_type(4)))  unsigned int u32x4;
typedef u32x4 __attribute__((may_alias)) u32x4a;

// Build a v16bf fragment from two 16-byte-aligned chunks.
static __device__ __forceinline__ v16bf load_frag2(const __bf16* p0, const __bf16* p1) {
  union { v16bf v; u32x4a q[2]; } u;
  u.q[0] = *(const u32x4a*)p0;
  u.q[1] = *(const u32x4a*)p1;
  return u.v;
}

// LDS byte address = low 32 bits of the flat shared-aperture address (ISA: LDS
// aperture maps addr[31:0] to the LDS offset).
static __device__ __forceinline__ unsigned lds_addr_of(const void* p) {
  return (unsigned)(size_t)p;
}

// Async global -> LDS copy, 4 bytes per lane (ISA 15.18.3 op 96, ASYNCcnt).
static __device__ __forceinline__ void async_g2l_b32(unsigned lds, const void* gptr) {
  asm volatile("global_load_async_to_lds_b32 %0, %1, off"
               :: "v"(lds), "v"(gptr) : "memory");
}
static __device__ __forceinline__ void wait_async0() {
  asm volatile("s_wait_asynccnt 0x0" ::: "memory");
}

// ---------------------------------------------------------------------------
// Shared bf16 GEMM core: block = 256 threads = 8 waves, block tile 128(M)x64(N),
// wave tile 16x64, K in chunks of 32, double-buffered LDS staging of the B
// chunk (transposed so each lane's B-fragment is two contiguous ds_load_b128).
// Staging: thread t loads 8 contiguous bf16 of row (t>>3) with one b128 load,
// scatters into LDS columns (t&7)*8..+7. A-fragment is software-pipelined.
// A layout per CDNA5 16-bit A spec: lane m=l&15, half=l>>4;
// k = {half*8..+7, 16+half*8..+7}.
// ---------------------------------------------------------------------------
static __device__ __forceinline__ void gemm_core(
    const __bf16* __restrict__ A, int lda,
    const __bf16* __restrict__ Bm, int ldb, int K,
    __bf16 (*Bt)[64][40], v8f* acc)
{
  const int tid  = threadIdx.x;
  const int lane = tid & 31, wave = tid >> 5, half = lane >> 4;
  const int mRow  = blockIdx.y * 128 + wave * 16 + (lane & 15);
  const int nBase = blockIdx.x * 64;
  const __bf16* arow = A + (size_t)mRow * lda;

  const int kk = tid >> 3, ng = (tid & 7) * 8;   // staging assignment
  auto stage = [&](int kc, int buf) {
    union { u32x4a q; __bf16 h[8]; } u;
    u.q = *(const u32x4a*)(Bm + (size_t)(kc + kk) * ldb + nBase + ng);
    if (kc + 32 < K)
      __builtin_prefetch(Bm + (size_t)(kc + 32 + kk) * ldb + nBase + ng, 0, 1);
    #pragma unroll
    for (int j = 0; j < 8; ++j) Bt[buf][ng + j][kk] = u.h[j];
  };

  stage(0, 0);
  v16bf a = load_frag2(arow + half * 8, arow + half * 8 + 16);
  __syncthreads();
  for (int kc = 0; kc < K; kc += 32) {
    const int cur = (kc >> 5) & 1;
    const bool hasNext = (kc + 32 < K);
    if (hasNext) stage(kc + 32, cur ^ 1);          // fill other buffer
    v16bf an = a;
    if (hasNext) {                                  // pipeline next A fragment
      const __bf16* ap = arow + (kc + 32) + half * 8;
      an = load_frag2(ap, ap + 16);
    }
    const int kb = half * 8;
    #pragma unroll
    for (int t = 0; t < 4; ++t) {
      const __bf16* bp = &Bt[cur][t * 16 + (lane & 15)][kb];
      const v16bf b = load_frag2(bp, bp + 16);
      acc[t] = __builtin_amdgcn_wmma_f32_16x16x32_bf16(
          false, a, false, b, (short)0, acc[t], false, false);
    }
    __syncthreads();
    a = an;
  }
}

// Plain epilogue: C_bf16[m*ldc+n] = alpha*acc (- 1 on diagonal if identSub).
__global__ __launch_bounds__(256) void gemm_bf16_plain(
    const __bf16* __restrict__ A, int lda,
    const __bf16* __restrict__ Bm, int ldb,
    __bf16* __restrict__ C, int ldc, int K, float alpha, int identSub)
{
  __shared__ alignas(16) __bf16 Bt[2][64][40];
  v8f acc[4] = {};
  gemm_core(A, lda, Bm, ldb, K, Bt, acc);
  const int lane = threadIdx.x & 31, wave = threadIdx.x >> 5, half = lane >> 4;
  const int mBase = blockIdx.y * 128 + wave * 16;
  const int nBase = blockIdx.x * 64;
  #pragma unroll
  for (int t = 0; t < 4; ++t) {
    const int nn = nBase + t * 16 + (lane & 15);
    #pragma unroll
    for (int r = 0; r < 8; ++r) {
      const int mm = mBase + r + half * 8;
      float v = alpha * acc[t][r];
      if (identSub && mm == nn) v -= 1.0f;
      C[(size_t)mm * ldc + nn] = (__bf16)v;
    }
  }
}

// Swizzled epilogue for per-node weight precompute: rows = node, col c = kidx*OUT+o.
// Stored so the per-node consumer reads each B fragment as lane-contiguous 32B:
// off = (node*KP + k32*32)*OUT + (o>>4)*512 + lane*16 + (kidx&15), lane=((kidx&31)>>4)*16+(o&15)
__global__ __launch_bounds__(256) void gemm_bf16_wswz(
    const __bf16* __restrict__ A, int lda,
    const __bf16* __restrict__ Bm, int ldb, int K,
    __bf16* __restrict__ W, int KP, int outShift)
{
  __shared__ alignas(16) __bf16 Bt[2][64][40];
  v8f acc[4] = {};
  gemm_core(A, lda, Bm, ldb, K, Bt, acc);
  const int lane = threadIdx.x & 31, wave = threadIdx.x >> 5, half = lane >> 4;
  const int mBase = blockIdx.y * 128 + wave * 16;
  const int nBase = blockIdx.x * 64;
  const int omask = (1 << outShift) - 1;
  #pragma unroll
  for (int t = 0; t < 4; ++t) {
    const int c = nBase + t * 16 + (lane & 15);
    const int kidx = c >> outShift;
    const int o = c & omask;
    #pragma unroll
    for (int r = 0; r < 8; ++r) {
      const int node = mBase + r + half * 8;
      size_t off = ((size_t)node * KP + (size_t)(kidx >> 5) * 32) << outShift;
      off += (size_t)(o >> 4) * 512
           + (size_t)((((kidx & 31) >> 4) << 4) + (o & 15)) * 16
           + (size_t)(kidx & 15);
      W[off] = (__bf16)acc[t][r];
    }
  }
}

// ---------------------------------------------------------------------------
// Per-node GEMMs: one workgroup per node, A = 16(batch) x KP gathered to LDS
// via async global->LDS copies (each (b,k)-segment is contiguous in both
// spaces); weights read in pre-swizzled fragment order.
// ---------------------------------------------------------------------------
template<int DI>
static __device__ __forceinline__ void gather_A(
    __bf16* Al, int n, int CPAD,
    const __bf16* __restrict__ Xc, const __bf16* __restrict__ XG1,
    const __bf16* __restrict__ XG2)
{
  constexpr int KP = ((3 * DI + 31) / 32) * 32;
  constexpr int DW = DI / 2;             // dwords per (b,k) segment
  const int tid = threadIdx.x;
  for (int u = tid; u < 16 * 3 * DW; u += 128) {
    const int seg = u / DW, dw = u - seg * DW;
    const int b = seg / 3, k = seg - (seg / 3) * 3;
    const __bf16* src = (k == 0) ? Xc : (k == 1) ? XG1 : XG2;
    const char* g = (const char*)(src + (size_t)n * CPAD + (size_t)b * DI) + (size_t)dw * 4;
    const unsigned l = lds_addr_of(&Al[b * KP + k * DI]) + (unsigned)dw * 4;
    async_g2l_b32(l, g);
  }
  if constexpr (KP != 3 * DI) {          // zero-fill padded K tail
    constexpr int TAIL = KP - 3 * DI;
    for (int u = tid; u < 16 * TAIL; u += 128) {
      const int b = u / TAIL, kq = 3 * DI + (u - (u / TAIL) * TAIL);
      Al[b * KP + kq] = (__bf16)0.0f;
    }
  }
  wait_async0();
  __syncthreads();
}

template<int DI>
__global__ __launch_bounds__(128) void pernode_gate_k(
    const __bf16* __restrict__ Xc, const __bf16* __restrict__ XG1,
    const __bf16* __restrict__ XG2, int CPAD,
    const __bf16* __restrict__ GW, const float* __restrict__ GB,
    float* __restrict__ ZR)
{
  constexpr int KP  = ((3 * DI + 31) / 32) * 32;
  constexpr int OUT = 128;
  __shared__ alignas(16) __bf16 Al[16 * KP];
  const int n = blockIdx.x, tid = threadIdx.x;
  gather_A<DI>(Al, n, CPAD, Xc, XG1, XG2);
  const int lane = tid & 31, wave = tid >> 5, half = lane >> 4;
  v8f acc[2] = {};
  const __bf16* arow = &Al[(lane & 15) * KP];
  for (int kc = 0; kc < KP; kc += 32) {
    const __bf16* ap = arow + kc + half * 16;
    const v16bf a = load_frag2(ap, ap + 8);
    const __bf16* wp = GW + (size_t)n * KP * OUT + (size_t)(kc >> 5) * (32 * OUT)
                          + (size_t)(wave * 2) * 512 + (size_t)lane * 16;
    #pragma unroll
    for (int t = 0; t < 2; ++t) {
      const v16bf b = load_frag2(wp + t * 512, wp + t * 512 + 8);
      acc[t] = __builtin_amdgcn_wmma_f32_16x16x32_bf16(
          false, a, false, b, (short)0, acc[t], false, false);
    }
  }
  const int ob = wave * 32 + (lane & 15);
  #pragma unroll
  for (int t = 0; t < 2; ++t) {
    const int o = ob + t * 16;
    #pragma unroll
    for (int r = 0; r < 8; ++r) {
      const int b = r + half * 8;
      float v = acc[t][r] + GB[n * OUT + o];
      v = 1.0f / (1.0f + __expf(-v));          // sigmoid -> z (o<64) / r (o>=64)
      ZR[((size_t)n * 16 + b) * 128 + o] = v;
    }
  }
}

template<int DI>
__global__ __launch_bounds__(128) void pernode_update_k(
    const __bf16* __restrict__ Xc, const __bf16* __restrict__ XG1,
    const __bf16* __restrict__ XG2, int CPAD,
    const __bf16* __restrict__ UW, const float* __restrict__ UB,
    const float* __restrict__ ZR, float* __restrict__ state,
    float* __restrict__ outT, long long outStrideB)
{
  constexpr int KP  = ((3 * DI + 31) / 32) * 32;
  constexpr int OUT = 64;
  __shared__ alignas(16) __bf16 Al[16 * KP];
  const int n = blockIdx.x, tid = threadIdx.x;
  gather_A<DI>(Al, n, CPAD, Xc, XG1, XG2);
  const int lane = tid & 31, wave = tid >> 5, half = lane >> 4;
  v8f acc = {};
  const __bf16* arow = &Al[(lane & 15) * KP];
  for (int kc = 0; kc < KP; kc += 32) {
    const __bf16* ap = arow + kc + half * 16;
    const v16bf a = load_frag2(ap, ap + 8);
    const __bf16* wp = UW + (size_t)n * KP * OUT + (size_t)(kc >> 5) * (32 * OUT)
                          + (size_t)wave * 512 + (size_t)lane * 16;
    const v16bf b = load_frag2(wp, wp + 8);
    acc = __builtin_amdgcn_wmma_f32_16x16x32_bf16(
        false, a, false, b, (short)0, acc, false, false);
  }
  const int o = wave * 16 + (lane & 15);
  #pragma unroll
  for (int r = 0; r < 8; ++r) {
    const int b = r + half * 8;
    const float hc = tanhf(acc[r] + UB[n * OUT + o]);
    const float rr = ZR[((size_t)n * 16 + b) * 128 + 64 + o];
    const size_t sidx = ((size_t)b * NODE + n) * HID + o;
    const float h = rr * state[sidx] + (1.0f - rr) * hc;
    state[sidx] = h;
    if (outT) outT[(size_t)b * outStrideB + (size_t)n * HID + o] = h;
  }
}

// ---------------------------------------------------------------------------
// Small helper kernels
// ---------------------------------------------------------------------------
__global__ __launch_bounds__(256) void row_softmax(
    const float* __restrict__ E, __bf16* __restrict__ Sb)
{
  const int n = blockIdx.x, tid = threadIdx.x;
  __shared__ float red[256];
  __shared__ float en[EDIM];
  if (tid < EDIM) en[tid] = E[n * EDIM + tid];
  __syncthreads();
  float vals[8];
  float mx = 0.0f;                       // relu >= 0
  #pragma unroll
  for (int c = 0; c < 8; ++c) {
    const int m = tid + c * 256;
    float d = 0.0f;
    #pragma unroll
    for (int dd = 0; dd < EDIM; ++dd) d += en[dd] * E[m * EDIM + dd];
    d = fmaxf(d, 0.0f);
    vals[c] = d;
    mx = fmaxf(mx, d);
  }
  red[tid] = mx; __syncthreads();
  for (int s = 128; s > 0; s >>= 1) { if (tid < s) red[tid] = fmaxf(red[tid], red[tid + s]); __syncthreads(); }
  mx = red[0]; __syncthreads();
  float sum = 0.0f;
  #pragma unroll
  for (int c = 0; c < 8; ++c) { vals[c] = __expf(vals[c] - mx); sum += vals[c]; }
  red[tid] = sum; __syncthreads();
  for (int s = 128; s > 0; s >>= 1) { if (tid < s) red[tid] += red[tid + s]; __syncthreads(); }
  const float inv = 1.0f / red[0];
  #pragma unroll
  for (int c = 0; c < 8; ++c)
    Sb[(size_t)n * NODE + tid + c * 256] = (__bf16)(vals[c] * inv);
}

__global__ __launch_bounds__(256) void conv_e_pad(
    const float* __restrict__ E, __bf16* __restrict__ Eb)
{
  const int idx = blockIdx.x * 256 + threadIdx.x;
  if (idx >= NODE * 32) return;
  const int n = idx >> 5, d = idx & 31;
  Eb[idx] = (d < EDIM) ? (__bf16)E[n * EDIM + d] : (__bf16)0.0f;
}

__global__ __launch_bounds__(256) void conv_w_pad(
    const float* __restrict__ W, __bf16* __restrict__ Wb, int C)
{
  const long long idx = (long long)blockIdx.x * 256 + threadIdx.x;
  if (idx >= (long long)32 * C) return;
  Wb[idx] = (idx < (long long)EDIM * C) ? (__bf16)W[idx] : (__bf16)0.0f;
}

__global__ __launch_bounds__(256) void bias_pre(
    const float* __restrict__ E, const float* __restrict__ Bp,
    float* __restrict__ outp, int O)
{
  const int idx = blockIdx.x * 256 + threadIdx.x;
  if (idx >= NODE * O) return;
  const int n = idx / O, o = idx - n * O;
  float s = 0.0f;
  #pragma unroll
  for (int d = 0; d < EDIM; ++d) s += E[n * EDIM + d] * Bp[d * O + o];
  outp[idx] = s;
}

// Xcat[m][b*DI+i] = i<DIN ? src1[b*s1b+m*s1m+i] : state[b][m][i-DIN] (* z if zr)
__global__ __launch_bounds__(256) void build_xcat(
    __bf16* __restrict__ Xc, int CPAD, int DI, int DIN,
    const float* __restrict__ src1, int s1b, int s1m,
    const float* __restrict__ state, const float* __restrict__ zr)
{
  const int m = blockIdx.y;
  const int c = blockIdx.x * 256 + threadIdx.x;
  if (c >= CPAD) return;
  float v = 0.0f;
  if (c < BB * DI) {
    const int b = c / DI, i = c - b * DI;
    if (i < DIN) {
      v = src1[(size_t)b * s1b + (size_t)m * s1m + i];
    } else {
      const int o = i - DIN;
      float s = state[((size_t)b * NODE + m) * HID + o];
      if (zr) s *= zr[((size_t)m * 16 + b) * 128 + o];
      v = s;
    }
  }
  Xc[(size_t)m * CPAD + c] = (__bf16)v;
}

__global__ __launch_bounds__(256) void init_states(
    const float* __restrict__ init, float* __restrict__ s0, float* __restrict__ s1, int n)
{
  const int i = blockIdx.x * 256 + threadIdx.x;
  if (i < n) { s0[i] = init[i]; s1[i] = init[n + i]; }
}

__global__ __launch_bounds__(256) void write_lasts(
    const float* __restrict__ s0, const float* __restrict__ s1, float* __restrict__ dst, int n)
{
  const int i = blockIdx.x * 256 + threadIdx.x;
  if (i < n) { dst[i] = s0[i]; dst[n + i] = s1[i]; }
}

// ---------------------------------------------------------------------------
extern "C" void kernel_launch(void* const* d_in, const int* in_sizes, int n_in,
                              void* d_out, int out_size, void* d_ws, size_t ws_size,
                              hipStream_t stream)
{
  (void)in_sizes; (void)n_in; (void)out_size; (void)ws_size;
  const float* x    = (const float*)d_in[0];
  const float* init = (const float*)d_in[1];
  const float* E    = (const float*)d_in[2];
  const float* gw0  = (const float*)d_in[3];
  const float* gb0  = (const float*)d_in[4];
  const float* uw0  = (const float*)d_in[5];
  const float* ub0  = (const float*)d_in[6];
  const float* gw1  = (const float*)d_in[7];
  const float* gb1  = (const float*)d_in[8];
  const float* uw1  = (const float*)d_in[9];
  const float* ub1  = (const float*)d_in[10];
  float* out = (float*)d_out;

  char* ws = (char*)d_ws;
  size_t off = 0;
  auto alloc = [&](size_t bytes) -> char* {
    char* p = ws + off;
    off = (off + bytes + 255) & ~(size_t)255;
    return p;
  };
  const size_t NN = (size_t)NODE * NODE;
  __bf16* Sb  = (__bf16*)alloc(NN * 2);
  __bf16* S2b = (__bf16*)alloc(NN * 2);
  __bf16* Eb  = (__bf16*)alloc((size_t)NODE * 32 * 2);
  __bf16* Wb  = (__bf16*)alloc((size_t)32 * 49152 * 2);
  __bf16* GW0 = (__bf16*)alloc((size_t)NODE * 224 * 128 * 2);
  __bf16* UW0 = (__bf16*)alloc((size_t)NODE * 224 * 64 * 2);
  __bf16* GW1 = (__bf16*)alloc((size_t)NODE * 384 * 128 * 2);
  __bf16* UW1 = (__bf16*)alloc((size_t)NODE * 384 * 64 * 2);
  float* GB0 = (float*)alloc((size_t)NODE * 128 * 4);
  float* UB0 = (float*)alloc((size_t)NODE * 64 * 4);
  float* GB1 = (float*)alloc((size_t)NODE * 128 * 4);
  float* UB1 = (float*)alloc((size_t)NODE * 64 * 4);
  __bf16* Xc  = (__bf16*)alloc(NN * 2);
  __bf16* XG1 = (__bf16*)alloc(NN * 2);
  __bf16* XG2 = (__bf16*)alloc(NN * 2);
  float* ZR  = (float*)alloc((size_t)NODE * 16 * 128 * 4);
  float* st0 = (float*)alloc((size_t)BB * NODE * HID * 4);
  float* st1 = (float*)alloc((size_t)BB * NODE * HID * 4);

  // ---- supports: S = softmax(relu(E E^T)); S2 = 2*S@S - I
  conv_e_pad<<<(NODE * 32 + 255) / 256, 256, 0, stream>>>(E, Eb);
  row_softmax<<<NODE, 256, 0, stream>>>(E, Sb);
  gemm_bf16_plain<<<dim3(32, 16), 256, 0, stream>>>(Sb, NODE, Sb, NODE, S2b, NODE, NODE, 2.0f, 1);

  // ---- per-node weights (E @ W, swizzled to WMMA B-fragment order) + biases
  const int C0g = 3 * 66 * 128, C0u = 3 * 66 * 64;
  const int C1g = 3 * 128 * 128, C1u = 3 * 128 * 64;
  conv_w_pad<<<(32 * C0g + 255) / 256, 256, 0, stream>>>(gw0, Wb, C0g);
  gemm_bf16_wswz<<<dim3(C0g / 64, 16), 256, 0, stream>>>(Eb, 32, Wb, C0g, 32, GW0, 224, 7);
  conv_w_pad<<<(32 * C0u + 255) / 256, 256, 0, stream>>>(uw0, Wb, C0u);
  gemm_bf16_wswz<<<dim3(C0u / 64, 16), 256, 0, stream>>>(Eb, 32, Wb, C0u, 32, UW0, 224, 6);
  conv_w_pad<<<(32 * C1g + 255) / 256, 256, 0, stream>>>(gw1, Wb, C1g);
  gemm_bf16_wswz<<<dim3(C1g / 64, 16), 256, 0, stream>>>(Eb, 32, Wb, C1g, 32, GW1, 384, 7);
  conv_w_pad<<<(32 * C1u + 255) / 256, 256, 0, stream>>>(uw1, Wb, C1u);
  gemm_bf16_wswz<<<dim3(C1u / 64, 16), 256, 0, stream>>>(Eb, 32, Wb, C1u, 32, UW1, 384, 6);
  bias_pre<<<(NODE * 128 + 255) / 256, 256, 0, stream>>>(E, gb0, GB0, 128);
  bias_pre<<<(NODE * 64 + 255) / 256, 256, 0, stream>>>(E, ub0, UB0, 64);
  bias_pre<<<(NODE * 128 + 255) / 256, 256, 0, stream>>>(E, gb1, GB1, 128);
  bias_pre<<<(NODE * 64 + 255) / 256, 256, 0, stream>>>(E, ub1, UB1, 64);

  const int BN = BB * NODE * HID;
  init_states<<<(BN + 255) / 256, 256, 0, stream>>>(init, st0, st1, BN);

  const int CP0 = 1088, CP1 = 2048;  // B*di padded to 64
  for (int t = 0; t < TT; ++t) {
    const float* xt = x + (size_t)t * NODE * 2;
    // ---- layer 0 (di = 2 + 64 = 66)
    build_xcat<<<dim3((CP0 + 255) / 256, NODE), 256, 0, stream>>>(
        Xc, CP0, 66, 2, xt, TT * NODE * 2, 2, st0, nullptr);
    gemm_bf16_plain<<<dim3(CP0 / 64, 16), 256, 0, stream>>>(Sb, NODE, Xc, CP0, XG1, CP0, NODE, 1.0f, 0);
    gemm_bf16_plain<<<dim3(CP0 / 64, 16), 256, 0, stream>>>(S2b, NODE, Xc, CP0, XG2, CP0, NODE, 1.0f, 0);
    pernode_gate_k<66><<<NODE, 128, 0, stream>>>(Xc, XG1, XG2, CP0, GW0, GB0, ZR);
    build_xcat<<<dim3((CP0 + 255) / 256, NODE), 256, 0, stream>>>(
        Xc, CP0, 66, 2, xt, TT * NODE * 2, 2, st0, ZR);
    gemm_bf16_plain<<<dim3(CP0 / 64, 16), 256, 0, stream>>>(Sb, NODE, Xc, CP0, XG1, CP0, NODE, 1.0f, 0);
    gemm_bf16_plain<<<dim3(CP0 / 64, 16), 256, 0, stream>>>(S2b, NODE, Xc, CP0, XG2, CP0, NODE, 1.0f, 0);
    pernode_update_k<66><<<NODE, 128, 0, stream>>>(Xc, XG1, XG2, CP0, UW0, UB0, ZR, st0, nullptr, 0);
    // ---- layer 1 (di = 64 + 64 = 128), input = layer-0 state (h0_t)
    build_xcat<<<dim3(CP1 / 256, NODE), 256, 0, stream>>>(
        Xc, CP1, 128, 64, st0, NODE * HID, HID, st1, nullptr);
    gemm_bf16_plain<<<dim3(CP1 / 64, 16), 256, 0, stream>>>(Sb, NODE, Xc, CP1, XG1, CP1, NODE, 1.0f, 0);
    gemm_bf16_plain<<<dim3(CP1 / 64, 16), 256, 0, stream>>>(S2b, NODE, Xc, CP1, XG2, CP1, NODE, 1.0f, 0);
    pernode_gate_k<128><<<NODE, 128, 0, stream>>>(Xc, XG1, XG2, CP1, GW1, GB1, ZR);
    build_xcat<<<dim3(CP1 / 256, NODE), 256, 0, stream>>>(
        Xc, CP1, 128, 64, st0, NODE * HID, HID, st1, ZR);
    gemm_bf16_plain<<<dim3(CP1 / 64, 16), 256, 0, stream>>>(Sb, NODE, Xc, CP1, XG1, CP1, NODE, 1.0f, 0);
    gemm_bf16_plain<<<dim3(CP1 / 64, 16), 256, 0, stream>>>(S2b, NODE, Xc, CP1, XG2, CP1, NODE, 1.0f, 0);
    pernode_update_k<128><<<NODE, 128, 0, stream>>>(
        Xc, XG1, XG2, CP1, UW1, UB1, ZR, st1,
        out + (size_t)t * NODE * HID, (long long)TT * NODE * HID);
  }
  write_lasts<<<(BN + 255) / 256, 256, 0, stream>>>(st0, st1, out + (size_t)BB * TT * NODE * HID, BN);
}
